// SAGE_5214090297412
// MI455X (gfx1250) — matrix operand order repaired and checked by reading, not compiled, not run on hardware
//
#include <hip/hip_runtime.h>

// ---------------------------------------------------------------------------
// GraphSAGE (GCN aggregator), 2 layers, for MI455X (gfx1250, wave32).
//   deg  = segment_sum(1, dst)
//   h1   = relu( ((scatter(x[src]->dst) + x) / (deg+1)) @ W1^T + b1 )
//   out  =       ((scatter(h1[src]->dst) + h1) / (deg+1)) @ W2^T + b2
// Roofline: GEMMs are ~5 GFLOP total (negligible); the edge gather/scatter is
// ~3.3 GB of random node-feature traffic, but all node arrays (51 MB each)
// fit in the 192 MB L2, so the bound is L2 atomic throughput. Aggregation is
// fp32 native global_atomic_add_f32 (inline asm, guaranteed no CAS loop);
// GEMMs are bf16 WMMA (v_wmma_f32_16x16x32_bf16) with fp32 accumulate.
// Workspace layout (floats): deg[N] | agg[N*128] | h1[N*128]  (~103 MB)
// ---------------------------------------------------------------------------

typedef __attribute__((ext_vector_type(16))) __bf16 v16bf;
typedef __attribute__((ext_vector_type(8)))  float  v8f;

struct B32x2 { uint4 lo; uint4 hi; };   // 32 bytes -> bit_cast to v16bf
static_assert(sizeof(B32x2) == 32, "frag pack size");

__device__ __forceinline__ unsigned short f32_to_bf16(float f) {
    unsigned int u = __float_as_uint(f);
    u += 0x7FFFu + ((u >> 16) & 1u);       // round-to-nearest-even
    return (unsigned short)(u >> 16);
}

__device__ __forceinline__ ushort4 pack_bf16x4(float a, float b, float c, float d) {
    ushort4 u;
    u.x = f32_to_bf16(a); u.y = f32_to_bf16(b);
    u.z = f32_to_bf16(c); u.w = f32_to_bf16(d);
    return u;
}

// Native CDNA5 fp32 atomic add, no return (STOREcnt path). Inline asm
// guarantees hardware lowering (no CAS-loop expansion by the atomics pass).
__device__ __forceinline__ void global_fadd(float* p, float v) {
    asm volatile("global_atomic_add_f32 %0, %1, off"
                 :: "v"(p), "v"(v) : "memory");
}

// --------------------------------- zero ------------------------------------
__global__ void zero_kernel(float* __restrict__ p, long long n) {
    const long long n4 = n >> 2;
    long long i = (long long)blockIdx.x * blockDim.x + threadIdx.x;
    const long long stride = (long long)gridDim.x * blockDim.x;
    float4* p4 = (float4*)p;
    const float4 z = make_float4(0.f, 0.f, 0.f, 0.f);
    for (long long j = i; j < n4; j += stride) p4[j] = z;
    if (i == 0)                                  // scalar tail (n % 4)
        for (long long j = n4 << 2; j < n; ++j) p[j] = 0.0f;
}

// ------------------------------- scatter -----------------------------------
// One wave per edge. 32 lanes x float4 = 128 features. Optionally builds deg.
template <bool BUILD_DEG>
__global__ __launch_bounds__(256)
void scatter_kernel(const float* __restrict__ feat,
                    const int*   __restrict__ src,
                    const int*   __restrict__ dst,
                    float*       __restrict__ agg,
                    float*       __restrict__ deg,
                    int n_edges) {
    const int wave = threadIdx.x >> 5;
    const int lane = threadIdx.x & 31;
    const int e = blockIdx.x * (blockDim.x >> 5) + wave;
    if (e >= n_edges) return;

    const int s = src[e];
    const int d = dst[e];

    const float4 v = *(const float4*)(feat + (size_t)s * 128 + lane * 4);
    float* ap = agg + (size_t)d * 128 + lane * 4;
    global_fadd(ap + 0, v.x);
    global_fadd(ap + 1, v.y);
    global_fadd(ap + 2, v.z);
    global_fadd(ap + 3, v.w);
    if (BUILD_DEG && lane == 0) global_fadd(&deg[d], 1.0f);
}

// ------------------------- fused SAGE layer (WMMA) -------------------------
// Block = 256 threads = 8 waves; covers 128 nodes (wave w -> rows 16w..16w+15).
// h_neigh = (agg + feat) / (deg + 1) staged into LDS as bf16, W staged as bf16.
// Each wave: D(16x16) += A(16x32) x B(32x16) over 4 K-steps per column tile.
template <int OUTF, bool RELU>
__global__ __launch_bounds__(256)
void sage_layer_kernel(const float* __restrict__ agg,
                       const float* __restrict__ feat,
                       const float* __restrict__ deg,
                       const float* __restrict__ W,     // [OUTF][128] row-major
                       const float* __restrict__ bias,  // [OUTF]
                       float*       __restrict__ out,   // [N][OUTF]
                       int n_nodes) {
    constexpr int K   = 128;
    constexpr int AST = 136;  // padded ushort row stride (272 B): no bank
                              // conflicts, 8B-aligned rows for b64 DS stores

    __shared__ unsigned short lds_w[OUTF * AST];
    __shared__ unsigned short lds_a[128 * AST];
    __shared__ float          lds_rinv[128];

    const int tid  = threadIdx.x;
    const int wave = tid >> 5;
    const int lane = tid & 31;
    const int base = blockIdx.x * 128;
    const bool full_block = (base + 128) <= n_nodes;

    // Per-row normalization factor (once per row, not per element).
    if (tid < 128) {
        const int node = base + tid;
        lds_rinv[tid] = (node < n_nodes) ? (1.0f / (deg[node] + 1.0f)) : 0.0f;
    }
    // Stage W as bf16: float4 global loads -> ushort4 (b64) LDS stores.
    for (int i = tid; i < (OUTF * K) / 4; i += 256) {
        const int idx4 = i * 4;
        const int j = idx4 >> 7, k = idx4 & 127;
        const float4 w = *(const float4*)(W + idx4);
        *(ushort4*)(&lds_w[j * AST + k]) = pack_bf16x4(w.x, w.y, w.z, w.w);
    }
    __syncthreads();
    // Stage normalized node features as bf16 (float4 loads, b64 DS stores).
    for (int i = tid; i < (128 * K) / 4; i += 256) {
        const int idx4 = i * 4;
        const int row = idx4 >> 7, k = idx4 & 127;
        const int node = base + row;
        ushort4 u = make_ushort4(0, 0, 0, 0);
        if (node < n_nodes) {
            const size_t off = (size_t)node * K + k;
            const float4 a = *(const float4*)(agg + off);
            const float4 f = *(const float4*)(feat + off);
            const float rinv = lds_rinv[row];
            u = pack_bf16x4((a.x + f.x) * rinv, (a.y + f.y) * rinv,
                            (a.z + f.z) * rinv, (a.w + f.w) * rinv);
        }
        *(ushort4*)(&lds_a[row * AST + k]) = u;
    }
    __syncthreads();

    const int m0   = wave * 16;
    const int half = lane >> 4;   // 0: lanes 0-15, 1: lanes 16-31
    const int r16  = lane & 15;

    // A fragments (ISA 16-bit A 16x32 layout):
    //   lanes 0-15 : elems 0..7 -> K=kb..kb+7,   elems 8..15 -> K=kb+16..kb+23
    //   lanes16-31 : elems 0..7 -> K=kb+8..+15,  elems 8..15 -> K=kb+24..+31
    v16bf afrag[4];
    {
        const unsigned short* Arow = &lds_a[(m0 + r16) * AST];
#pragma unroll
        for (int ks = 0; ks < 4; ++ks) {
            const int kb = ks * 32 + half * 8;
            B32x2 p;
            p.lo = *(const uint4*)(Arow + kb);
            p.hi = *(const uint4*)(Arow + kb + 16);
            afrag[ks] = __builtin_bit_cast(v16bf, p);
        }
    }

#pragma unroll
    for (int c = 0; c < OUTF / 16; ++c) {
        const float bb = bias[c * 16 + r16];  // D col = lane&15 for every VGPR
        v8f acc;
#pragma unroll
        for (int r = 0; r < 8; ++r) acc[r] = bb;

        // B fragments (32x16, col = lane&15, same K layout as A fragments).
        const unsigned short* Brow = &lds_w[(c * 16 + r16) * AST];
#pragma unroll
        for (int ks = 0; ks < 4; ++ks) {
            const int kb = ks * 32 + half * 8;
            B32x2 p;
            p.lo = *(const uint4*)(Brow + kb);
            p.hi = *(const uint4*)(Brow + kb + 16);
            const v16bf bfrag = __builtin_bit_cast(v16bf, p);
            acc = __builtin_amdgcn_wmma_f32_16x16x32_bf16(
                false, afrag[ks], false, bfrag, (short)0, acc, false, false);
        }

        // D layout: VGPR r -> M = r (lanes 0-15) / r+8 (lanes 16-31).
        float* orow = out + (size_t)(base + m0 + half * 8) * OUTF + c * 16 + r16;
        if (full_block) {
            // Block-uniform fast path: no per-element exec masking.
#pragma unroll
            for (int r = 0; r < 8; ++r) {
                float v = acc[r];
                if (RELU) v = fmaxf(v, 0.0f);
                orow[(size_t)r * OUTF] = v;
            }
        } else {
#pragma unroll
            for (int r = 0; r < 8; ++r) {
                const int node = base + m0 + half * 8 + r;
                float v = acc[r];
                if (RELU) v = fmaxf(v, 0.0f);
                if (node < n_nodes) orow[(size_t)r * OUTF] = v;
            }
        }
    }
}

// --------------------------------- launch ----------------------------------
extern "C" void kernel_launch(void* const* d_in, const int* in_sizes, int n_in,
                              void* d_out, int out_size, void* d_ws, size_t ws_size,
                              hipStream_t stream) {
    const float* x   = (const float*)d_in[0];
    const int*   src = (const int*)  d_in[1];
    const int*   dst = (const int*)  d_in[2];
    const float* W1  = (const float*)d_in[3];
    const float* b1  = (const float*)d_in[4];
    const float* W2  = (const float*)d_in[5];
    const float* b2  = (const float*)d_in[6];

    const int N = in_sizes[0] / 128;
    const int E = in_sizes[1];

    // Workspace: deg[N] | agg[N*128] | h1[N*128]   (~103 MB for N=100k)
    float* deg = (float*)d_ws;
    float* agg = deg + N;                 // N*4 = 400000 B, 16B-aligned
    float* h1  = agg + (size_t)N * 128;

    const int edge_blocks = (E + 7) / 8;        // 8 edges (waves) per block
    const int node_blocks = (N + 127) / 128;    // 128 nodes per block

    // 1) zero deg + agg (contiguous region)
    zero_kernel<<<4096, 256, 0, stream>>>(deg, (long long)N * 129);
    // 2) layer-1 scatter (also builds deg)
    scatter_kernel<true><<<edge_blocks, 256, 0, stream>>>(x, src, dst, agg, deg, E);
    // 3) layer 1: norm + GEMM(128x128) + bias + relu -> h1
    sage_layer_kernel<128, true><<<node_blocks, 256, 0, stream>>>(
        agg, x, deg, W1, b1, h1, N);
    // 4) re-zero agg for layer 2
    zero_kernel<<<4096, 256, 0, stream>>>(agg, (long long)N * 128);
    // 5) layer-2 scatter of h1
    scatter_kernel<false><<<edge_blocks, 256, 0, stream>>>(h1, src, dst, agg, nullptr, E);
    // 6) layer 2: norm + GEMM(128x64) + bias -> d_out
    sage_layer_kernel<64, false><<<node_blocks, 256, 0, stream>>>(
        agg, h1, deg, W2, b2, (float*)d_out, N);
}
